// SMoE_MHA_36661840839515
// MI455X (gfx1250) — matrix operand order repaired and checked by reading, not compiled, not run on hardware
//
#include <hip/hip_runtime.h>
#include <math.h>

// ---------------------------------------------------------------------------
// Types for CDNA5 WMMA (wave32): A/B = 16 bf16 per lane, C/D = 8 f32 per lane
// ---------------------------------------------------------------------------
typedef __attribute__((ext_vector_type(16))) __bf16        v16bf;
typedef __attribute__((ext_vector_type(8)))  float         v8f;
typedef __attribute__((ext_vector_type(4)))  unsigned int  v4u;
typedef __attribute__((ext_vector_type(8)))  int           v8i;
typedef __attribute__((ext_vector_type(4)))  int           v4i;

union FragU { v4u q[2]; v16bf bf; };

__device__ __forceinline__ unsigned short f2bf(float f) {
  unsigned int u = __float_as_uint(f);
  unsigned int r = (u + 0x7FFFu + ((u >> 16) & 1u)) >> 16;
  return (unsigned short)r;
}

__device__ __forceinline__ unsigned int pk2bf(float lo, float hi) {
  return (unsigned int)f2bf(lo) | ((unsigned int)f2bf(hi) << 16);
}

__device__ __forceinline__ v8f zero8() {
  v8f z = {0.f, 0.f, 0.f, 0.f, 0.f, 0.f, 0.f, 0.f};
  return z;
}

// A-matrix 16x32 bf16 fragment (ISA 7.12.2): lane row m = lane&15,
// lo lanes hold K = [0..7] and [16..23], hi lanes +8. Two 16B loads.
__device__ __forceinline__ v16bf load_a_frag(const unsigned short* row, int hi) {
  FragU f;
  f.q[0] = *(const v4u*)(row + 8 * hi);
  f.q[1] = *(const v4u*)(row + 16 + 8 * hi);
  return f.bf;
}

// B-matrix 32x16 bf16 fragment: lane col n = lane&15, lo lanes K=[0..7]
// and [16..23], hi lanes +8 (matches A). Two 16B loads.
__device__ __forceinline__ v16bf load_b_frag(const unsigned short* colp, int hi) {
  FragU f;
  f.q[0] = *(const v4u*)(colp + 8 * hi);
  f.q[1] = *(const v4u*)(colp + 16 + 8 * hi);
  return f.bf;
}

__device__ __forceinline__ v8f wmma_bf16(v16bf a, v16bf b, v8f c) {
  return __builtin_amdgcn_wmma_f32_16x16x32_bf16(false, a, false, b,
                                                 (short)0, c, false, false);
}

// Async global->LDS 16B copy (GLOBAL_LOAD_ASYNC_TO_LDS_B128, ASYNCcnt path).
__device__ __forceinline__ void async_copy16(unsigned int ldsOff,
                                             const void* gptr) {
  asm volatile("global_load_async_to_lds_b128 %0, %1, off"
               :: "v"(ldsOff), "v"((unsigned long long)(size_t)gptr)
               : "memory");
}

__device__ __forceinline__ void async_wait0() {
  asm volatile("s_wait_asynccnt 0x0" ::: "memory");
}

__device__ __forceinline__ unsigned int lds_off(const void* p) {
  return (unsigned int)(size_t)p;
}

// ---------------------------------------------------------------------------
// TDM bulk copy: 64KB contiguous global -> LDS via tensor_load_to_lds.
// D# per ISA 8.3/8.4: 1D tile of 8192 x 8-byte elements.
// 6-arg builtin variant: (g0, g1, g2, g3, g_extra, cpol).
// ---------------------------------------------------------------------------
#if __has_builtin(__builtin_amdgcn_tensor_load_to_lds)
#define USE_TDM 1
__device__ __forceinline__ void tdm_copy64k(unsigned int ldsOff,
                                            const void* gptr) {
  unsigned long long ga = (unsigned long long)(size_t)gptr;
  // group0: count=1 | lds_addr | global_addr[56:0] | type=2 (bits 127:126)
  v4u g0 = {1u, ldsOff, (unsigned int)ga,
            ((unsigned int)(ga >> 32) & 0x01FFFFFFu) | 0x80000000u};
  // group1: data_size=3 (8B); tensor_dim0=8192 (bits 79:48);
  // tensor_dim1=1 (bits 111:80); tile_dim0=8192 (bits 127:112);
  // tile_dim1=1 (bits 143:128); tensor_dim0_stride=8192 (bits 207:160)
  v8i g1 = {0x00030000, 0x20000000, 0x00010000, 0x20000000,
            1, 8192, 0, 0};
  v4i z4 = {0, 0, 0, 0};
  v8i z8 = {0, 0, 0, 0, 0, 0, 0, 0};
  __builtin_amdgcn_tensor_load_to_lds(g0, g1, z4, z4, z8, 0);
}
#endif

// ---------------------------------------------------------------------------
// K1a: f32 -> bf16 cast with transpose.
// src: [R][C] row-major (batched, stride R*C). dst: [C][R] (n-major).
// ---------------------------------------------------------------------------
__global__ void k_cast_T(const float* __restrict__ src,
                         unsigned short* __restrict__ dst, int R, int C) {
  int i = blockIdx.x * blockDim.x + threadIdx.x;  // over R*C
  int b = blockIdx.y;
  if (i < R * C) {
    int n = i / R;
    int k = i - n * R;
    dst[(size_t)b * R * C + (size_t)n * R + k] =
        f2bf(src[(size_t)b * R * C + (size_t)k * C + n]);
  }
}

// ---------------------------------------------------------------------------
// K2: xp = x @ Wp + bp, fused head-split. 128x128 tile / block, 8 waves.
// B tile staged with async global->LDS (Wp pre-transposed to [n][k]);
// A tile: float2 loads, packed 2xbf16 ds_store_b32.
// ---------------------------------------------------------------------------
__global__ __launch_bounds__(256) void k_proj_split(
    const float* __restrict__ X, const unsigned short* __restrict__ WpT,
    const float* __restrict__ bp, float* __restrict__ xf32,
    unsigned short* __restrict__ xfb) {
  __shared__ __align__(16) unsigned short sA[128 * 32];   // [m][k]
  __shared__ __align__(16) unsigned short sBT[128 * 32];  // [n][k]
  const int tid = threadIdx.x;
  const int lane = tid & 31, wid = tid >> 5;
  const int hi = lane >> 4, ln = lane & 15;
  const int t0 = blockIdx.x * 128;
  const int n0 = blockIdx.y * 128;
  const int wm = wid >> 2, wn = wid & 3;  // 2 x 4 wave grid
  const unsigned int sB0 = lds_off(&sBT[0]);

  v8f acc[4][2];
  for (int i = 0; i < 4; i++)
    for (int j = 0; j < 2; j++) acc[i][j] = zero8();

  for (int k0 = 0; k0 < 1024; k0 += 32) {
    // stage B tile via async copy: rows n, 64B contiguous (k-major)
    for (int i = 0; i < 2; i++) {
      int c = tid * 2 + i;  // 512 chunks of 16B
      int row = c >> 2, ch = c & 3;
      async_copy16(sB0 + (unsigned int)((row * 32 + ch * 8) * 2),
                   WpT + (size_t)(n0 + row) * 1024 + k0 + ch * 8);
    }
    // stage A tile: float2 -> packed 2xbf16 -> ds_store_b32
    for (int rep = 0; rep < 8; rep++) {
      int r = (tid >> 4) + 16 * rep;
      int kp = (tid & 15) * 2;
      float2 v = *(const float2*)(X + (size_t)(t0 + r) * 1024 + k0 + kp);
      *(unsigned int*)(sA + r * 32 + kp) = pk2bf(v.x, v.y);
    }
    // prefetch next A k-slab (global_prefetch_b8)
    if (k0 + 32 < 1024)
      __builtin_prefetch(X + (size_t)(t0 + (tid >> 4)) * 1024 + k0 + 32 +
                             (tid & 15) * 2, 0, 1);
    async_wait0();
    __syncthreads();
    v16bf af[4], bfr[2];
    for (int i = 0; i < 4; i++)
      af[i] = load_a_frag(sA + (wm * 64 + i * 16 + ln) * 32, hi);
    for (int j = 0; j < 2; j++)
      bfr[j] = load_b_frag(sBT + (wn * 32 + j * 16 + ln) * 32, hi);
    for (int i = 0; i < 4; i++)
      for (int j = 0; j < 2; j++) acc[i][j] = wmma_bf16(af[i], bfr[j], acc[i][j]);
    __syncthreads();
  }

  // epilogue: bias + head split
  for (int i = 0; i < 4; i++)
    for (int j = 0; j < 2; j++) {
      int d = n0 + wn * 32 + j * 16 + ln;
      float bb = bp[d];
      int head = d >> 7, dd = d & 127;
      for (int c = 0; c < 8; c++) {
        int t = t0 + wm * 64 + i * 16 + c + 8 * hi;
        int b = t >> 10, s = t & 1023;
        size_t idx = ((size_t)(b * 8 + head) * 1024 + s) * 128 + dd;
        float v = acc[i][j][c] + bb;
        xf32[idx] = v;
        xfb[idx] = f2bf(v);
      }
    }
}

// ---------------------------------------------------------------------------
// K3: gating -- scores, softmax, top-2, re-softmax weights.
// ---------------------------------------------------------------------------
__global__ __launch_bounds__(256) void k_gating(
    const float* __restrict__ xf32, const float* __restrict__ Wg,
    const float* __restrict__ bg, float* __restrict__ gate,
    float* __restrict__ tscore, float* __restrict__ tidx,
    float* __restrict__ wdense) {
  int t = blockIdx.x * blockDim.x + threadIdx.x;  // 0..65535
  const float4* xr = (const float4*)(xf32 + (size_t)t * 128);
  float sc[8];
  for (int e = 0; e < 8; e++) sc[e] = bg[e];
  for (int q = 0; q < 32; q++) {
    float4 v = xr[q];
    const float* wrow = Wg + (q * 4) * 8;  // Wg is [128][8]
    for (int e = 0; e < 8; e++)
      sc[e] += v.x * wrow[e] + v.y * wrow[8 + e] + v.z * wrow[16 + e] +
               v.w * wrow[24 + e];
  }
  float mx = sc[0];
  for (int e = 1; e < 8; e++) mx = fmaxf(mx, sc[e]);
  float den = 0.f, p[8];
  for (int e = 0; e < 8; e++) { p[e] = __expf(sc[e] - mx); den += p[e]; }
  float inv = 1.f / den;
  for (int e = 0; e < 8; e++) {
    p[e] *= inv;
    gate[(size_t)t * 8 + e] = sc[e];
  }
  int i0 = 0;
  for (int e = 1; e < 8; e++) if (p[e] > p[i0]) i0 = e;
  int i1 = (i0 == 0) ? 1 : 0;
  for (int e = 0; e < 8; e++) if (e != i0 && p[e] > p[i1]) i1 = e;
  float p0 = p[i0], p1 = p[i1];
  float m2 = fmaxf(p0, p1);
  float e0 = __expf(p0 - m2), e1 = __expf(p1 - m2);
  float w0 = e0 / (e0 + e1), w1 = e1 / (e0 + e1);
  tscore[(size_t)t * 2 + 0] = p0;
  tscore[(size_t)t * 2 + 1] = p1;
  tidx[(size_t)t * 2 + 0] = (float)i0;
  tidx[(size_t)t * 2 + 1] = (float)i1;
  float* wd = wdense + (size_t)t * 8;
  for (int e = 0; e < 8; e++) wd[e] = 0.f;
  wd[i0] = w0;
  wd[i1] = w1;
}

// ---------------------------------------------------------------------------
// K4: fused MoE FFN. Block = 128 tokens (8 waves x 16), loops all 8 experts.
// W1[e]^T / W2[e]^T (64KB each, pre-transposed) staged via TDM
// tensor_load_to_lds (TENSORcnt path); async-copy fallback.
// ---------------------------------------------------------------------------
__global__ __launch_bounds__(256) void k_moe_ffn(
    const unsigned short* __restrict__ xfb, const float* __restrict__ xf32,
    const unsigned short* __restrict__ W1T, const float* __restrict__ b1,
    const unsigned short* __restrict__ W2T, const float* __restrict__ b2,
    const float* __restrict__ wdense, unsigned short* __restrict__ xcomb) {
  extern __shared__ unsigned short smem[];
  unsigned short* sW1T = smem;           // [256][128]  (n-major)
  unsigned short* sW2T = smem + 32768;   // [128][256]  (n-major)
  unsigned short* sH1  = smem + 65536;   // 8 waves x [16][256]
  const int tid = threadIdx.x;
  const int lane = tid & 31, wid = tid >> 5;
  const int hi = lane >> 4, ln = lane & 15;
  const int wtok = blockIdx.x * 128 + wid * 16;  // flattened (n*1024+s)
  const unsigned int sW1o = lds_off(sW1T);
  const unsigned int sW2o = lds_off(sW2T);

  // A fragments for GEMM1: x tile [16][128] bf16, all 4 k-steps
  v16bf af[4];
  for (int kk = 0; kk < 4; kk++)
    af[kk] = load_a_frag(xfb + (size_t)(wtok + ln) * 128 + kk * 32, hi);

  v8f comb[8];
  for (int j = 0; j < 8; j++) comb[j] = zero8();

  for (int e = 0; e < 8; e++) {
    const unsigned short* w1 = W1T + (size_t)e * 32768;
    const unsigned short* w2 = W2T + (size_t)e * 32768;
#ifdef USE_TDM
    if (wid == 0) {
      tdm_copy64k(sW1o, w1);
      tdm_copy64k(sW2o, w2);
      __builtin_amdgcn_s_wait_tensorcnt(0);
    }
#else
    for (int rep = 0; rep < 16; rep++) {
      int c = tid + rep * 256;  // 4096 chunks of 16B each
      async_copy16(sW1o + (unsigned int)c * 16u, w1 + c * 8);
      async_copy16(sW2o + (unsigned int)c * 16u, w2 + c * 8);
    }
    async_wait0();
#endif
    __syncthreads();

    // per-row top-2 gate weight for this expert (0 if not selected)
    float wv[8];
    for (int c = 0; c < 8; c++)
      wv[c] = wdense[(size_t)(wtok + c + 8 * hi) * 8 + e];

    // GEMM1: h1 = x @ W1[e] + b1[e], erf-GELU, -> LDS (A-layout for GEMM2)
    unsigned short* h1w = sH1 + wid * 4096;
    for (int ns = 0; ns < 16; ns++) {
      v8f h = zero8();
      for (int kk = 0; kk < 4; kk++) {
        v16bf bfr = load_b_frag(sW1T + (ns * 16 + ln) * 128 + kk * 32, hi);
        h = wmma_bf16(af[kk], bfr, h);
      }
      int col = ns * 16 + ln;
      float bb = b1[e * 256 + col];
      for (int c = 0; c < 8; c++) {
        float xv = h[c] + bb;
        float g = 0.5f * xv * (1.f + erff(xv * 0.70710678118f));
        h1w[(c + 8 * hi) * 256 + col] = f2bf(g);
      }
    }

    // GEMM2: eo = gelu(h1) @ W2[e] + b2[e]; weighted accumulate
    for (int ns = 0; ns < 8; ns++) {
      v8f eo = zero8();
      for (int kk = 0; kk < 8; kk++) {
        v16bf a2 = load_a_frag(h1w + ln * 256 + kk * 32, hi);
        v16bf b2f = load_b_frag(sW2T + (ns * 16 + ln) * 256 + kk * 32, hi);
        eo = wmma_bf16(a2, b2f, eo);
      }
      int col = ns * 16 + ln;
      float bb = b2[e * 128 + col];
      for (int c = 0; c < 8; c++) comb[ns][c] += wv[c] * (eo[c] + bb);
    }
    __syncthreads();
  }

  // residual + merge heads -> merged bf16 [t][d]
  for (int ns = 0; ns < 8; ns++) {
    int col = ns * 16 + ln;
    for (int c = 0; c < 8; c++) {
      int tok = wtok + c + 8 * hi;
      float v = comb[ns][c] + xf32[(size_t)tok * 128 + col];
      int n = tok >> 10, s = tok & 1023;
      int b = n >> 3, head = n & 7;
      xcomb[((size_t)(b * 1024 + s)) * 1024 + head * 128 + col] = f2bf(v);
    }
  }
}

// ---------------------------------------------------------------------------
// K5: out = merged @ Wa + ba. Both A and B tiles staged via async copies.
// ---------------------------------------------------------------------------
__global__ __launch_bounds__(256) void k_out_proj(
    const unsigned short* __restrict__ A, const unsigned short* __restrict__ WaT,
    const float* __restrict__ ba, float* __restrict__ out) {
  __shared__ __align__(16) unsigned short sA[128 * 32];   // [m][k]
  __shared__ __align__(16) unsigned short sBT[128 * 32];  // [n][k]
  const int tid = threadIdx.x;
  const int lane = tid & 31, wid = tid >> 5;
  const int hi = lane >> 4, ln = lane & 15;
  const int t0 = blockIdx.x * 128;
  const int n0 = blockIdx.y * 128;
  const int wm = wid >> 2, wn = wid & 3;
  const unsigned int sA0 = lds_off(&sA[0]);
  const unsigned int sB0 = lds_off(&sBT[0]);

  v8f acc[4][2];
  for (int i = 0; i < 4; i++)
    for (int j = 0; j < 2; j++) acc[i][j] = zero8();

  for (int k0 = 0; k0 < 1024; k0 += 32) {
    for (int i = 0; i < 2; i++) {
      int c = tid * 2 + i;  // 512 chunks of 16B
      int row = c >> 2, ch = c & 3;
      unsigned int loff = (unsigned int)((row * 32 + ch * 8) * 2);
      async_copy16(sA0 + loff, A + (size_t)(t0 + row) * 1024 + k0 + ch * 8);
      async_copy16(sB0 + loff, WaT + (size_t)(n0 + row) * 1024 + k0 + ch * 8);
    }
    async_wait0();
    __syncthreads();
    v16bf af[4], bfr[2];
    for (int i = 0; i < 4; i++)
      af[i] = load_a_frag(sA + (wm * 64 + i * 16 + ln) * 32, hi);
    for (int j = 0; j < 2; j++)
      bfr[j] = load_b_frag(sBT + (wn * 32 + j * 16 + ln) * 32, hi);
    for (int i = 0; i < 4; i++)
      for (int j = 0; j < 2; j++) acc[i][j] = wmma_bf16(af[i], bfr[j], acc[i][j]);
    __syncthreads();
  }

  for (int i = 0; i < 4; i++)
    for (int j = 0; j < 2; j++) {
      int d = n0 + wn * 32 + j * 16 + ln;
      float bb = ba[d];
      for (int c = 0; c < 8; c++) {
        int t = t0 + wm * 64 + i * 16 + c + 8 * hi;
        out[(size_t)t * 1024 + d] = acc[i][j][c] + bb;
      }
    }
}

// ---------------------------------------------------------------------------
// Host launch
// ---------------------------------------------------------------------------
extern "C" void kernel_launch(void* const* d_in, const int* in_sizes, int n_in,
                              void* d_out, int out_size, void* d_ws,
                              size_t ws_size, hipStream_t stream) {
  const float* x  = (const float*)d_in[0];
  const float* Wp = (const float*)d_in[1];
  const float* bp = (const float*)d_in[2];
  const float* Wg = (const float*)d_in[3];
  const float* bg = (const float*)d_in[4];
  const float* W1 = (const float*)d_in[5];
  const float* b1 = (const float*)d_in[6];
  const float* W2 = (const float*)d_in[7];
  const float* b2 = (const float*)d_in[8];
  const float* Wa = (const float*)d_in[9];
  const float* ba = (const float*)d_in[10];

  char* ws = (char*)d_ws;
  unsigned short* wsWpT  = (unsigned short*)(ws);             // 2 MB  [n][k]
  unsigned short* wsWaT  = (unsigned short*)(ws + 2097152);   // 2 MB  [n][k]
  unsigned short* wsW1T  = (unsigned short*)(ws + 4194304);   // 512KB [e][n][k]
  unsigned short* wsW2T  = (unsigned short*)(ws + 4718592);   // 512KB [e][n][k]
  float*          xf32   = (float*)(ws + 5242880);            // 32 MB
  unsigned short* xfb    = (unsigned short*)(ws + 38797312);  // 16 MB
  unsigned short* xcomb  = (unsigned short*)(ws + 55574528);  // 16 MB
  float*          wdense = (float*)(ws + 72351744);           // 2 MB

  float* out  = (float*)d_out;               // [8,1024,1024]
  float* o_ts = out + 8388608;               // topk_scores [64,1024,2]
  float* o_ti = o_ts + 131072;               // topk_idx    [64,1024,2]
  float* o_g  = o_ti + 131072;               // gate_scores [64,1024,8]

  // transpose-casts: Wp[k][n]->WpT[n][k], Wa likewise, W1[e],W2[e] likewise
  k_cast_T<<<dim3(4096, 1), 256, 0, stream>>>(Wp, wsWpT, 1024, 1024);
  k_cast_T<<<dim3(4096, 1), 256, 0, stream>>>(Wa, wsWaT, 1024, 1024);
  k_cast_T<<<dim3(128, 8), 256, 0, stream>>>(W1, wsW1T, 128, 256);
  k_cast_T<<<dim3(128, 8), 256, 0, stream>>>(W2, wsW2T, 256, 128);

  dim3 gGemm(64, 8);
  k_proj_split<<<gGemm, 256, 0, stream>>>(x, wsWpT, bp, xf32, xfb);
  k_gating<<<256, 256, 0, stream>>>(xf32, Wg, bg, o_g, o_ts, o_ti, wdense);
  k_moe_ffn<<<512, 256, 196608, stream>>>(xfb, xf32, wsW1T, b1, wsW2T, b2,
                                          wdense, xcomb);
  k_out_proj<<<gGemm, 256, 0, stream>>>(xcomb, wsWaT, ba, out);
}
